// HeterEdgeDenseGAT_35562329211433
// MI455X (gfx1250) — matrix operand order repaired and checked by reading, not compiled, not run on hardware
//
#include <hip/hip_runtime.h>
#include <hip/hip_bf16.h>
#include <stdint.h>

#define BSZ   2
#define NADJ  2
#define NTOK  2048
#define DEMB  64
#define NFEAT 192
#define NH    8
#define F1    64
#define RH    (NADJ*NH)   /* 16 */

typedef __attribute__((ext_vector_type(16))) __bf16 bf16x16;
typedef __attribute__((ext_vector_type(8)))  __bf16 bf16x8;
typedef __attribute__((ext_vector_type(8)))  float  f32x8;

static __device__ __forceinline__ unsigned short f2bfbits(float f) {
  unsigned int u = __builtin_bit_cast(unsigned int, f);
  unsigned int r = u + 0x7fffu + ((u >> 16) & 1u);
  return (unsigned short)(r >> 16);
}
static __device__ __forceinline__ __bf16 f2bf(float f) {
  unsigned short h = f2bfbits(f);
  return __builtin_bit_cast(__bf16, h);
}

// ---------------------------------------------------------------------------
// feat = concat(emb0[v], emb1[v], local_emb)  -> f32 + bf16 copies
// ---------------------------------------------------------------------------
__global__ __launch_bounds__(256) void k_feat(
    const int* __restrict__ verts, const float* __restrict__ local_emb,
    const float* __restrict__ emb0, const float* __restrict__ emb1,
    float* __restrict__ featf, unsigned short* __restrict__ featb)
{
  int idx = blockIdx.x * 256 + threadIdx.x;
  if (idx >= BSZ * NTOK * NFEAT) return;
  int f = idx % NFEAT; int bn = idx / NFEAT;
  float v;
  if (f < DEMB)          v = emb0[(size_t)verts[bn] * DEMB + f];
  else if (f < 2 * DEMB) v = emb1[(size_t)verts[bn] * DEMB + (f - DEMB)];
  else                   v = local_emb[(size_t)bn * DEMB + (f - 2 * DEMB)];
  featf[idx] = v;
  featb[idx] = f2bfbits(v);
}

// ---------------------------------------------------------------------------
// w (RH, K, 64) f32 -> wT (RH, 64, K) bf16  (contiguous B-operand rows)
// ---------------------------------------------------------------------------
__global__ __launch_bounds__(256) void k_prep_w(
    const float* __restrict__ w, unsigned short* __restrict__ wT, int K)
{
  int idx = blockIdx.x * 256 + threadIdx.x;
  if (idx >= RH * K * F1) return;
  int o = idx % F1; int rem = idx / F1; int k = rem % K; int rh = rem / K;
  wT[((size_t)rh * F1 + o) * K + k] = f2bfbits(w[idx]);
}

// ---------------------------------------------------------------------------
// hp = A(bf16, MxK) @ W^T via v_wmma_f32_16x16x32_bf16.  One wave computes a
// 32x64 tile (two 16-row A tiles share each B tile).  Stores hp transposed
// bf16 (hpT[64][M]) for the attention GEMM, plus s = tanh(hp)@a_src and
// d = tanh(hp)@a_dst (f32, 16-lane shfl reduced).
// A batch index = b*(RH/aDiv) + rh/aDiv.
// ---------------------------------------------------------------------------
__global__ __launch_bounds__(32) void k_hp(
    const unsigned short* __restrict__ A,   // [Abat][M][K] bf16
    const unsigned short* __restrict__ WT,  // [RH][F1][K] bf16
    const float* __restrict__ a_src,        // [RH][F1]
    const float* __restrict__ a_dst,        // [RH][F1]
    unsigned short* __restrict__ hpT,       // [BSZ][RH][F1][M] bf16
    float* __restrict__ s_out,              // [BSZ][RH][M]
    float* __restrict__ d_out,              // [BSZ][RH][M]
    int M, int K, int aDiv)
{
  const int lane = threadIdx.x;
  const int mt = blockIdx.x, rh = blockIdx.y, b = blockIdx.z;
  const int ln = lane & 15, hi = lane >> 4;
  const int kh8 = hi * 8, kh16 = hi * 16;
  const int aIdx = b * (RH / aDiv) + rh / aDiv;
  const unsigned short* Arow0 = A + ((size_t)aIdx * M + mt * 32 + ln) * K;
  const unsigned short* Arow1 = Arow0 + (size_t)16 * K;
  const unsigned short* Wrh   = WT + (size_t)rh * F1 * K;

  f32x8 cc[2][4];
#pragma unroll
  for (int t = 0; t < 2; t++)
#pragma unroll
    for (int nt = 0; nt < 4; nt++)
#pragma unroll
      for (int j = 0; j < 8; j++) cc[t][nt][j] = 0.f;

  for (int kb = 0; kb < K; kb += 32) {
    bf16x8 lo0  = *(const bf16x8*)(Arow0 + kb + kh16);
    bf16x8 hi0  = *(const bf16x8*)(Arow0 + kb + kh16 + 8);
    bf16x8 lo1  = *(const bf16x8*)(Arow1 + kb + kh16);
    bf16x8 hi1  = *(const bf16x8*)(Arow1 + kb + kh16 + 8);
    bf16x16 av0, av1;
#pragma unroll
    for (int e = 0; e < 8; e++) {
      av0[e] = lo0[e]; av0[8 + e] = hi0[e];
      av1[e] = lo1[e]; av1[8 + e] = hi1[e];
    }
#pragma unroll
    for (int nt = 0; nt < 4; nt++) {
      bf16x16 bv = *(const bf16x16*)(Wrh + ((size_t)(nt * 16 + ln)) * K + kb + kh16);
      cc[0][nt] = __builtin_amdgcn_wmma_f32_16x16x32_bf16(
          false, av0, false, bv, (short)0, cc[0][nt], false, false);
      cc[1][nt] = __builtin_amdgcn_wmma_f32_16x16x32_bf16(
          false, av1, false, bv, (short)0, cc[1][nt], false, false);
    }
  }

  const size_t hb = ((size_t)b * RH + rh) * (size_t)F1 * M;
  const size_t sb = ((size_t)b * RH + rh) * (size_t)M;
  float sa[2][8], da[2][8];
#pragma unroll
  for (int t = 0; t < 2; t++)
#pragma unroll
    for (int j = 0; j < 8; j++) { sa[t][j] = 0.f; da[t][j] = 0.f; }

#pragma unroll
  for (int nt = 0; nt < 4; nt++) {
    int col = nt * 16 + ln;
    float ws = a_src[rh * F1 + col], wd = a_dst[rh * F1 + col];
#pragma unroll
    for (int t = 0; t < 2; t++) {
#pragma unroll
      for (int j = 0; j < 8; j++) {
        float v = cc[t][nt][j];
        int m = mt * 32 + t * 16 + j + 8 * hi;
        hpT[hb + (size_t)col * M + m] = f2bfbits(v);
        float th = tanhf(v);
        sa[t][j] += th * ws; da[t][j] += th * wd;
      }
    }
  }
#pragma unroll
  for (int t = 0; t < 2; t++) {
#pragma unroll
    for (int j = 0; j < 8; j++) {
      float sv = sa[t][j], dv = da[t][j];
      for (int off = 8; off > 0; off >>= 1) {
        sv += __shfl_xor(sv, off, 16);
        dv += __shfl_xor(dv, off, 16);
      }
      if (ln == 0) {
        int m = mt * 32 + t * 16 + j + 8 * hi;
        s_out[sb + m] = sv; d_out[sb + m] = dv;
      }
    }
  }
}

// ---------------------------------------------------------------------------
// Two-pass masked-softmax row statistics.  Logits are rank-1 + mask and are
// recomputed (never materialized): l = leakyrelu(s[row]+d[col]), adj>0.
// ---------------------------------------------------------------------------
__global__ __launch_bounds__(256) void k_rowstats(
    const float* __restrict__ adj, const float* __restrict__ s,
    const float* __restrict__ dvec, float* __restrict__ rmax, float* __restrict__ rsum)
{
  __shared__ float red[256];
  const int row = blockIdx.x, rh = blockIdx.y, b = blockIdx.z, tid = threadIdx.x;
  const int r = rh >> 3;
  const float* arow = adj + (((size_t)b * NADJ + r) * NTOK + row) * (size_t)NTOK;
  const size_t sb = ((size_t)b * RH + rh) * (size_t)NTOK;
  const float srow = s[sb + row];
  const float* dp = dvec + sb;

  float mx = -3.4e38f;
  for (int c = tid; c < NTOK; c += 256) {
    float a = arow[c];
    if (a > 0.f) { float x = srow + dp[c]; float l = x > 0.f ? x : 0.2f * x; mx = fmaxf(mx, l); }
  }
  red[tid] = mx; __syncthreads();
  for (int st = 128; st > 0; st >>= 1) { if (tid < st) red[tid] = fmaxf(red[tid], red[tid + st]); __syncthreads(); }
  const float m = red[0]; __syncthreads();

  float sum = 0.f;
  for (int c = tid; c < NTOK; c += 256) {
    float a = arow[c];
    if (a > 0.f) { float x = srow + dp[c]; float l = x > 0.f ? x : 0.2f * x; sum += __expf(l - m); }
  }
  red[tid] = sum; __syncthreads();
  for (int st = 128; st > 0; st >>= 1) { if (tid < st) red[tid] += red[tid + st]; __syncthreads(); }
  if (tid == 0) { rmax[sb + row] = m; rsum[sb + row] = red[0]; }
}

// ---------------------------------------------------------------------------
// out[32x64] = softmax(mask(leakyrelu(s+d))) @ hp.  P tiles are regenerated
// in-register directly into the bf16 A-operand layout (two 16-row groups
// share the B tiles and the d[] loads); hpT supplies B tiles as contiguous
// 32B loads.  Unnormalized accumulate, divide by rowsum at the end.
// ---------------------------------------------------------------------------
__global__ __launch_bounds__(32) void k_attn(
    const float* __restrict__ adj, const float* __restrict__ s,
    const float* __restrict__ dvec, const float* __restrict__ rmax,
    const float* __restrict__ rsum, const unsigned short* __restrict__ hpT,
    float* __restrict__ out)
{
  const int lane = threadIdx.x;
  const int mt = blockIdx.x, rh = blockIdx.y, b = blockIdx.z;
  const int ln = lane & 15, hi = lane >> 4;
  const int kh8 = hi * 8, kh16 = hi * 16;
  const int r = rh >> 3;
  const int ma0 = mt * 32 + ln;
  const int ma1 = ma0 + 16;
  const size_t sb = ((size_t)b * RH + rh) * (size_t)NTOK;
  const float srow0 = s[sb + ma0], srow1 = s[sb + ma1];
  const float rmx0  = rmax[sb + ma0], rmx1 = rmax[sb + ma1];
  const float* arow0 = adj + (((size_t)b * NADJ + r) * NTOK + ma0) * (size_t)NTOK;
  const float* arow1 = arow0 + (size_t)16 * NTOK;
  const float* dp = dvec + sb;
  const unsigned short* hB = hpT + ((size_t)b * RH + rh) * (size_t)F1 * NTOK;

  f32x8 cc[2][4];
#pragma unroll
  for (int t = 0; t < 2; t++)
#pragma unroll
    for (int nt = 0; nt < 4; nt++)
#pragma unroll
      for (int j = 0; j < 8; j++) cc[t][nt][j] = 0.f;

  for (int kb = 0; kb < NTOK; kb += 32) {
    if (kb + 64 < NTOK) {
      __builtin_prefetch(arow0 + kb + 64, 0, 1);
      __builtin_prefetch(arow1 + kb + 64, 0, 1);
    }
    bf16x16 pv0, pv1;
#pragma unroll
    for (int e = 0; e < 8; e++) {
      float dc0 = dp[kb + kh16 + e];
      float dc1 = dp[kb + kh16 + 8 + e];
      {
        float a0 = arow0[kb + kh16 + e];
        float x0 = srow0 + dc0; float l0 = x0 > 0.f ? x0 : 0.2f * x0;
        pv0[e] = f2bf(a0 > 0.f ? __expf(l0 - rmx0) : 0.f);
        float a1 = arow0[kb + kh16 + 8 + e];
        float x1 = srow0 + dc1; float l1 = x1 > 0.f ? x1 : 0.2f * x1;
        pv0[8 + e] = f2bf(a1 > 0.f ? __expf(l1 - rmx0) : 0.f);
      }
      {
        float a0 = arow1[kb + kh16 + e];
        float x0 = srow1 + dc0; float l0 = x0 > 0.f ? x0 : 0.2f * x0;
        pv1[e] = f2bf(a0 > 0.f ? __expf(l0 - rmx1) : 0.f);
        float a1 = arow1[kb + kh16 + 8 + e];
        float x1 = srow1 + dc1; float l1 = x1 > 0.f ? x1 : 0.2f * x1;
        pv1[8 + e] = f2bf(a1 > 0.f ? __expf(l1 - rmx1) : 0.f);
      }
    }
#pragma unroll
    for (int nt = 0; nt < 4; nt++) {
      bf16x16 bv = *(const bf16x16*)(hB + ((size_t)(nt * 16 + ln)) * NTOK + kb + kh16);
      cc[0][nt] = __builtin_amdgcn_wmma_f32_16x16x32_bf16(
          false, pv0, false, bv, (short)0, cc[0][nt], false, false);
      cc[1][nt] = __builtin_amdgcn_wmma_f32_16x16x32_bf16(
          false, pv1, false, bv, (short)0, cc[1][nt], false, false);
    }
  }

  float inv[2][8];
#pragma unroll
  for (int t = 0; t < 2; t++)
#pragma unroll
    for (int j = 0; j < 8; j++)
      inv[t][j] = 1.f / rsum[sb + mt * 32 + t * 16 + j + 8 * hi];
  float* ob = out + ((size_t)b * RH + rh) * (size_t)NTOK * F1;
#pragma unroll
  for (int nt = 0; nt < 4; nt++) {
    int col = nt * 16 + ln;
#pragma unroll
    for (int t = 0; t < 2; t++)
#pragma unroll
      for (int j = 0; j < 8; j++)
        ob[(size_t)(mt * 32 + t * 16 + j + 8 * hi) * F1 + col] = cc[t][nt][j] * inv[t][j];
  }
}

// ---------------------------------------------------------------------------
// x1[b][r][n][h*64+o] = elu(h0[b][r*8+h][n][o])  -> f32 + bf16
// ---------------------------------------------------------------------------
__global__ __launch_bounds__(256) void k_x1(
    const float* __restrict__ h0, float* __restrict__ x1f, unsigned short* __restrict__ x1b)
{
  int idx = blockIdx.x * 256 + threadIdx.x;
  if (idx >= BSZ * NADJ * NTOK * NH * F1) return;
  int o = idx % F1; int rem = idx / F1;
  int h = rem % NH; rem /= NH;
  int n = rem % NTOK; rem /= NTOK;
  int r = rem % NADJ; int b = rem / NADJ;
  float v = h0[(((size_t)b * RH + r * NH + h) * NTOK + n) * F1 + o];
  float e = v > 0.f ? v : (__expf(v) - 1.f);
  x1f[idx] = e;
  x1b[idx] = f2bfbits(e);
}

// ---------------------------------------------------------------------------
// mean over heads
// ---------------------------------------------------------------------------
__global__ __launch_bounds__(256) void k_mean(
    const float* __restrict__ h1h, float* __restrict__ h1)
{
  int idx = blockIdx.x * 256 + threadIdx.x;
  if (idx >= BSZ * NADJ * NTOK * F1) return;
  int o = idx % F1; int rem = idx / F1;
  int n = rem % NTOK; rem /= NTOK;
  int r = rem % NADJ; int b = rem / NADJ;
  float acc = 0.f;
#pragma unroll
  for (int h = 0; h < NH; h++)
    acc += h1h[(((size_t)b * RH + r * NH + h) * NTOK + n) * F1 + o];
  h1[idx] = acc * (1.f / NH);
}

// ---------------------------------------------------------------------------
// fw1 = feat @ aw1 (kept f32; tiny)
// ---------------------------------------------------------------------------
__global__ __launch_bounds__(256) void k_fw1(
    const float* __restrict__ featf, const float* __restrict__ aw1, float* __restrict__ fw1)
{
  int idx = blockIdx.x * 256 + threadIdx.x;
  if (idx >= BSZ * NTOK * F1) return;
  int o = idx % F1; int bn = idx / F1;
  const float* fr = featf + (size_t)bn * NFEAT;
  float acc = 0.f;
  for (int f = 0; f < NFEAT; f++) acc += fr[f] * aw1[(size_t)f * F1 + o];
  fw1[idx] = acc;
}

// ---------------------------------------------------------------------------
// fusion head + log-softmax.  64 threads per (b,n).
// ---------------------------------------------------------------------------
__global__ __launch_bounds__(64) void k_final(
    const float* __restrict__ fw1, const float* __restrict__ h1,
    const float* __restrict__ aw2, const float* __restrict__ am, float* __restrict__ out)
{
  __shared__ float red[64];
  __shared__ float sc[NADJ];
  __shared__ float sm1;
  const int bn = blockIdx.x; const int o = threadIdx.x;
  const int b = bn / NTOK, n = bn % NTOK;
  const float fw = fw1[(size_t)bn * F1 + o];

  for (int r = 0; r < NADJ; r++) {
    const float* hr = h1 + (((size_t)b * NADJ + r) * NTOK + n) * F1;
    float acc = 0.f;
    for (int f = 0; f < F1; f++) acc += hr[f] * aw2[f * F1 + o];
    float q = tanhf(fw + acc);
    red[o] = q * am[o]; __syncthreads();
    for (int st = 32; st > 0; st >>= 1) { if (o < st) red[o] += red[o + st]; __syncthreads(); }
    if (o == 0) sc[r] = red[0];
    __syncthreads();
  }
  float m = fmaxf(sc[0], sc[1]);
  float e0 = __expf(sc[0] - m), e1 = __expf(sc[1] - m);
  float inv = 1.f / (e0 + e1);
  float b0 = e0 * inv, b1 = e1 * inv;
  const float* hr0 = h1 + (((size_t)b * NADJ + 0) * NTOK + n) * F1;
  const float* hr1 = h1 + (((size_t)b * NADJ + 1) * NTOK + n) * F1;
  float fused = b0 * hr0[o] + b1 * hr1[o];

  red[o] = fused; __syncthreads();
  for (int st = 32; st > 0; st >>= 1) { if (o < st) red[o] = fmaxf(red[o], red[o + st]); __syncthreads(); }
  if (o == 0) sm1 = red[0];
  __syncthreads();
  float mx = sm1;
  red[o] = __expf(fused - mx); __syncthreads();
  for (int st = 32; st > 0; st >>= 1) { if (o < st) red[o] += red[o + st]; __syncthreads(); }
  if (o == 0) sm1 = red[0];
  __syncthreads();
  out[(size_t)bn * F1 + o] = fused - mx - __logf(sm1);
}

// ---------------------------------------------------------------------------
extern "C" void kernel_launch(void* const* d_in, const int* in_sizes, int n_in,
                              void* d_out, int out_size, void* d_ws, size_t ws_size,
                              hipStream_t stream) {
  const float* adj      = (const float*)d_in[0];
  const int*   verts    = (const int*)  d_in[1];
  const float* lemb     = (const float*)d_in[2];
  const float* emb0     = (const float*)d_in[3];
  const float* emb1     = (const float*)d_in[4];
  const float* w_l0     = (const float*)d_in[5];
  const float* a_src_l0 = (const float*)d_in[6];
  const float* a_dst_l0 = (const float*)d_in[7];
  const float* w_l1     = (const float*)d_in[8];
  const float* a_src_l1 = (const float*)d_in[9];
  const float* a_dst_l1 = (const float*)d_in[10];
  const float* aw1      = (const float*)d_in[11];
  const float* aw2      = (const float*)d_in[12];
  const float* am       = (const float*)d_in[13];
  float* outp = (float*)d_out;

  char* W = (char*)d_ws;
  size_t off = 0;
  auto carve = [&](size_t bytes) -> void* {
    void* p = W + off;
    off = (off + bytes + 255) & ~(size_t)255;
    return p;
  };
  float*          featf = (float*)         carve((size_t)BSZ*NTOK*NFEAT*4);
  unsigned short* featb = (unsigned short*)carve((size_t)BSZ*NTOK*NFEAT*2);
  unsigned short* w0T   = (unsigned short*)carve((size_t)RH*NFEAT*F1*2);
  unsigned short* w1T   = (unsigned short*)carve((size_t)RH*(NH*F1)*F1*2);
  unsigned short* hpT   = (unsigned short*)carve((size_t)BSZ*RH*F1*NTOK*2);
  float*          sbuf  = (float*)carve((size_t)BSZ*RH*NTOK*4);
  float*          dbuf  = (float*)carve((size_t)BSZ*RH*NTOK*4);
  float*          rmx   = (float*)carve((size_t)BSZ*RH*NTOK*4);
  float*          rsm   = (float*)carve((size_t)BSZ*RH*NTOK*4);
  float*          h0    = (float*)carve((size_t)BSZ*RH*NTOK*F1*4);  // reused as h1h
  float*          x1f   = (float*)carve((size_t)BSZ*NADJ*NTOK*(NH*F1)*4);
  unsigned short* x1b   = (unsigned short*)carve((size_t)BSZ*NADJ*NTOK*(NH*F1)*2);
  float*          h1    = (float*)carve((size_t)BSZ*NADJ*NTOK*F1*4);
  float*          fw1p  = (float*)carve((size_t)BSZ*NTOK*F1*4);
  (void)ws_size; (void)in_sizes; (void)n_in; (void)out_size;

  const dim3 gWave(NTOK / 32, RH, BSZ);   // one wave per 32x64 tile
  const dim3 gRow(NTOK, RH, BSZ);

  // feat
  k_feat<<<(BSZ*NTOK*NFEAT + 255)/256, 256, 0, stream>>>(verts, lemb, emb0, emb1, featf, featb);
  // weight transpose+cvt
  k_prep_w<<<(RH*NFEAT*F1 + 255)/256, 256, 0, stream>>>(w_l0, w0T, NFEAT);
  k_prep_w<<<(RH*(NH*F1)*F1 + 255)/256, 256, 0, stream>>>(w_l1, w1T, NH*F1);

  // ---- layer 0 ----
  k_hp<<<gWave, 32, 0, stream>>>(featb, w0T, a_src_l0, a_dst_l0, hpT, sbuf, dbuf,
                                 NTOK, NFEAT, RH /* A batch = b */);
  k_rowstats<<<gRow, 256, 0, stream>>>(adj, sbuf, dbuf, rmx, rsm);
  k_attn<<<gWave, 32, 0, stream>>>(adj, sbuf, dbuf, rmx, rsm, hpT, h0);
  k_x1<<<(BSZ*NADJ*NTOK*NH*F1 + 255)/256, 256, 0, stream>>>(h0, x1f, x1b);

  // ---- layer 1 ----
  k_hp<<<gWave, 32, 0, stream>>>(x1b, w1T, a_src_l1, a_dst_l1, hpT, sbuf, dbuf,
                                 NTOK, NH*F1, NH /* A batch = b*NADJ + r */);
  k_rowstats<<<gRow, 256, 0, stream>>>(adj, sbuf, dbuf, rmx, rsm);
  k_attn<<<gWave, 32, 0, stream>>>(adj, sbuf, dbuf, rmx, rsm, hpT, h0 /* h1h */);
  k_mean<<<(BSZ*NADJ*NTOK*F1 + 255)/256, 256, 0, stream>>>(h0, h1);

  // ---- fusion head ----
  k_fw1<<<(BSZ*NTOK*F1 + 255)/256, 256, 0, stream>>>(featf, aw1, fw1p);
  k_final<<<BSZ*NTOK, 64, 0, stream>>>(fw1p, h1, aw2, am, outp);
}
